// ECPGLinear_15058155339849
// MI455X (gfx1250) — matrix operand order repaired and checked by reading, not compiled, not run on hardware
//
#include <hip/hip_runtime.h>
#include <hip/hip_bf16.h>

typedef __attribute__((ext_vector_type(16))) _Float16 v16h;
typedef __attribute__((ext_vector_type(8)))  _Float16 v8h;
typedef __attribute__((ext_vector_type(8)))  float    v8f;
typedef __attribute__((ext_vector_type(4)))  float    v4f;
typedef __attribute__((ext_vector_type(4)))  int      v4i;

namespace {
constexpr int kM = 8192;          // 4 * 2048 rows of x
constexpr int kN = 4096;          // out features
constexpr int kK = 4096;          // in features
constexpr int BM = 256;           // block rows (x)
constexpr int BN = 128;           // block cols (out features)
constexpr int BK = 32;            // one WMMA K-step per LDS stage
constexpr int LDA = BK + 8;       // padded half-stride: 80B rows, 16B-aligned chunks
constexpr int LDB = BK + 8;
constexpr int KT  = kK / BK;      // 128 K-iterations
constexpr int GROUPS_PER_ROW = kK / 128;  // 32 scale groups per output row
}

__global__ __launch_bounds__(256) void ternary_wmma_gemm(
    const float* __restrict__ x,
    const int*   __restrict__ tern,
    const float* __restrict__ scales,
    float*       __restrict__ out)
{
  __shared__ _Float16 sA[2][BM * LDA];   // x tile, f16, [m][k]      (40 KB)
  __shared__ _Float16 sB[2][BN * LDB];   // dequantized W, f16, [n][k] (20 KB)

  const int tid  = threadIdx.x;
  const int lane = tid & 31;
  const int wave = tid >> 5;
  const int wm   = wave >> 1;            // 0..3 : 64-row slab
  const int wn   = wave & 1;             // 0..1 : 64-col slab

  const int m0 = blockIdx.y * BM;
  const int n0 = blockIdx.x * BN;

  // cooperative loader coords: 256 threads, 32 rows x (8 x float4) per pass
  const int lrow = tid >> 3;             // 0..31
  const int lcol = (tid & 7) * 4;        // 0,4,...,28

  // ---- load one K-tile: x fp32 -> f16 LDS; ternary int32 * group-scale -> f16 LDS
  auto load_tile = [&](int kt, int buf) {
    const int k0 = kt * BK;
    #pragma unroll
    for (int p = 0; p < BM / 32; ++p) {        // 8 passes for 256 rows
      const int r = lrow + p * 32;
      const v4f xv = *(const v4f*)(x + (size_t)(m0 + r) * kK + k0 + lcol);
      _Float16* dst = &sA[buf][r * LDA + lcol];
      dst[0] = (_Float16)xv.x; dst[1] = (_Float16)xv.y;
      dst[2] = (_Float16)xv.z; dst[3] = (_Float16)xv.w;
    }
    // BK=32 tile always lies inside one 128-wide scale group
    const int gk = k0 >> 7;
    #pragma unroll
    for (int p = 0; p < BN / 32; ++p) {        // 4 passes for 128 rows
      const int r = lrow + p * 32;
      const int n = n0 + r;
      const float s = scales[n * GROUPS_PER_ROW + gk];
      const v4i tv = *(const v4i*)(tern + (size_t)n * kK + k0 + lcol);
      _Float16* dst = &sB[buf][r * LDB + lcol];
      dst[0] = (_Float16)((float)tv.x * s);
      dst[1] = (_Float16)((float)tv.y * s);
      dst[2] = (_Float16)((float)tv.z * s);
      dst[3] = (_Float16)((float)tv.w * s);
    }
  };

  // fragment lane geometry (ISA 7.12.2, 16-bit A 16x32 / B 32x16, wave32)
  const int hrow  = lane & 15;
  const int koffA = (lane >= 16) ? 8  : 0;  // A: K chunks at koffA and koffA+16
  const int koffB = (lane >= 16) ? 8  : 0;  // B: K chunks at koffB and koffB+16

  v8f acc[4][4] = {};

  load_tile(0, 0);
  __syncthreads();

  int buf = 0;
  for (int kt = 0; kt < KT; ++kt) {
    if (kt + 1 < KT) load_tile(kt + 1, buf ^ 1);   // fill other buffer while computing

    v16h afrag[4], bfrag[4];
    #pragma unroll
    for (int i = 0; i < 4; ++i) {
      const int row = wm * 64 + i * 16 + hrow;
      const v8h lo = *(const v8h*)&sA[buf][row * LDA + koffA];
      const v8h hi = *(const v8h*)&sA[buf][row * LDA + koffA + 16];
      #pragma unroll
      for (int e = 0; e < 8; ++e) { afrag[i][e] = lo[e]; afrag[i][e + 8] = hi[e]; }
    }
    #pragma unroll
    for (int j = 0; j < 4; ++j) {
      const int row = wn * 64 + j * 16 + hrow;     // B col n == W row n, K contiguous
      const v8h lo = *(const v8h*)&sB[buf][row * LDB + koffB];
      const v8h hi = *(const v8h*)&sB[buf][row * LDB + koffB + 16];
      #pragma unroll
      for (int e = 0; e < 8; ++e) { bfrag[j][e] = lo[e]; bfrag[j][e + 8] = hi[e]; }
    }

    #pragma unroll
    for (int i = 0; i < 4; ++i)
      #pragma unroll
      for (int j = 0; j < 4; ++j)
        acc[i][j] = __builtin_amdgcn_wmma_f32_16x16x32_f16(
            /*neg_a=*/false, afrag[i], /*neg_b=*/false, bfrag[j],
            /*c_mod=*/(short)0, acc[i][j], /*reuse_a=*/false, /*reuse_b=*/false);

    __syncthreads();
    buf ^= 1;
  }

  // C/D layout: VGPR r -> M = r + (lane>=16 ? 8 : 0), N = lane & 15
  const int mrow = (lane >= 16) ? 8 : 0;
  #pragma unroll
  for (int i = 0; i < 4; ++i) {
    const int mbase = m0 + wm * 64 + i * 16 + mrow;
    #pragma unroll
    for (int j = 0; j < 4; ++j) {
      const int ng = n0 + wn * 64 + j * 16 + (lane & 15);
      #pragma unroll
      for (int r = 0; r < 8; ++r)
        out[(size_t)(mbase + r) * kN + ng] = acc[i][j][r];
    }
  }
}

extern "C" void kernel_launch(void* const* d_in, const int* in_sizes, int n_in,
                              void* d_out, int out_size, void* d_ws, size_t ws_size,
                              hipStream_t stream) {
  const float* x      = (const float*)d_in[0];
  const int*   tern   = (const int*)d_in[1];
  const float* scales = (const float*)d_in[2];
  float* out = (float*)d_out;

  dim3 grid(kN / BN, kM / BM);   // (32, 32) blocks, 256 threads = 8 waves each
  ternary_wmma_gemm<<<grid, dim3(256), 0, stream>>>(x, tern, scales, out);
}
